// SelfAttention_74483322847631
// MI455X (gfx1250) — matrix operand order repaired and checked by reading, not compiled, not run on hardware
//
#include <hip/hip_runtime.h>

typedef __bf16 bf16;
typedef bf16  bf16x8 __attribute__((ext_vector_type(8)));
typedef bf16  v16bf  __attribute__((ext_vector_type(16)));
typedef float v8f    __attribute__((ext_vector_type(8)));
typedef float f32x8  __attribute__((ext_vector_type(8)));
typedef int   v4i    __attribute__((ext_vector_type(4)));

#define CCH  128
#define NPOS 4096
#define TOK  16384   // 4 * 4096
#define NT   (NPOS / 32)

__device__ __forceinline__ v8f wmma_bf16(v16bf a, v16bf b, v8f c) {
  return __builtin_amdgcn_wmma_f32_16x16x32_bf16(
      /*neg_a=*/false, a, /*neg_b=*/false, b,
      /*c_mod=*/(short)0, c, /*reuse_a=*/false, /*reuse_b=*/false);
}

// ---------------- CDNA5 async global->LDS copy (ASYNCcnt) -------------------
typedef __attribute__((address_space(1))) v4i* as1_v4i_ptr;
typedef __attribute__((address_space(3))) v4i* as3_v4i_ptr;

#if __has_builtin(__builtin_amdgcn_global_load_async_to_lds_b128)
__device__ __forceinline__ void async_copy16(const bf16* g, bf16* l) {
  __builtin_amdgcn_global_load_async_to_lds_b128(
      (as1_v4i_ptr)(g), (as3_v4i_ptr)(l), 0, 0);
}
template <int N> __device__ __forceinline__ void tile_wait() {
#if __has_builtin(__builtin_amdgcn_s_wait_asynccnt)
  __builtin_amdgcn_s_wait_asynccnt(N);
#else
  asm volatile("s_wait_asynccnt %0" ::"i"(N) : "memory");
#endif
}
#else
// fallback: synchronous 16B copy through VGPRs (load + ds_store)
__device__ __forceinline__ void async_copy16(const bf16* g, bf16* l) {
  *(v4i*)l = *(const v4i*)g;
}
template <int N> __device__ __forceinline__ void tile_wait() {
  asm volatile("s_wait_dscnt 0" ::: "memory");
}
#endif

// A fragment (16x32 bf16, M x K): lane = row (lr), K chunks {hi*8.., 16+hi*8..}
__device__ __forceinline__ v16bf frag_A_bf16(const bf16* p, int ld, int row,
                                             int cb, int lr, int hi) {
  const bf16* r = p + (size_t)(row + lr) * ld + cb;
  bf16x8 c0 = *(const bf16x8*)(r + hi * 8);
  bf16x8 c1 = *(const bf16x8*)(r + 16 + hi * 8);
  v16bf a;
#pragma unroll
  for (int e = 0; e < 8; ++e) { a[e] = c0[e]; a[e + 8] = c1[e]; }
  return a;
}

// A fragment sourced from f32 memory (converted to bf16)
__device__ __forceinline__ v16bf frag_A_f32(const float* p, int ld, int row,
                                            int cb, int lr, int hi) {
  const float* r = p + (size_t)(row + lr) * ld + cb;
  f32x8 c0 = *(const f32x8*)(r + hi * 8);
  f32x8 c1 = *(const f32x8*)(r + 16 + hi * 8);
  v16bf a;
#pragma unroll
  for (int e = 0; e < 8; ++e) { a[e] = (bf16)c0[e]; a[e + 8] = (bf16)c1[e]; }
  return a;
}

// B fragment (32x16 bf16, K x N): lane = column, K chunks {hi*8.., 16+hi*8..} at kb.
// p points to [col][K] row-major storage.
__device__ __forceinline__ v16bf frag_B_bf16(const bf16* p, int ld, int col0,
                                             int kb, int lc, int hi) {
  const bf16* r = p + (size_t)(col0 + lc) * ld + kb;
  bf16x8 c0 = *(const bf16x8*)(r + hi * 8);
  bf16x8 c1 = *(const bf16x8*)(r + 16 + hi * 8);
  v16bf b;
#pragma unroll
  for (int e = 0; e < 8; ++e) { b[e] = c0[e]; b[e + 8] = c1[e]; }
  return b;
}

// ---------------- Kernel 0: transpose + convert weights to bf16 -------------
__global__ __launch_bounds__(256) void prep_weights(
    const float* __restrict__ Wq, const float* __restrict__ Wk,
    const float* __restrict__ Wv, bf16* __restrict__ WqT,
    bf16* __restrict__ WkT, bf16* __restrict__ WvT) {
  int id = blockIdx.x * 256 + threadIdx.x;
  if (id >= 3 * CCH * CCH) return;
  int w = id >> 14, e = id & (CCH * CCH - 1);
  int cin = e >> 7, cout = e & 127;
  const float* src = (w == 0) ? Wq : (w == 1) ? Wk : Wv;
  bf16* dst = (w == 0) ? WqT : (w == 1) ? WkT : WvT;
  dst[cout * CCH + cin] = (bf16)src[cin * CCH + cout];
}

// ---------------- Kernel 1: QKV projection (WMMA) ---------------------------
__global__ __launch_bounds__(128) void qkv_proj(
    const float* __restrict__ x, const float* __restrict__ bq,
    const float* __restrict__ bk, const float* __restrict__ bv,
    const bf16* __restrict__ WqT, const bf16* __restrict__ WkT,
    const bf16* __restrict__ WvT, bf16* __restrict__ qs,
    bf16* __restrict__ kbuf, bf16* __restrict__ vT) {
  const int lane = threadIdx.x & 31, wave = threadIdx.x >> 5;
  const int lr = lane & 15, hi = lane >> 4;
  const int r0 = (blockIdx.x * 4 + wave) * 16;
  const int b = r0 >> 12, rb = r0 & (NPOS - 1);

  v16bf xa[4];
#pragma unroll
  for (int c = 0; c < 4; ++c) xa[c] = frag_A_f32(x, CCH, r0, 32 * c, lr, hi);

  const bf16* Ws[3] = {WqT, WkT, WvT};
  const float* bias[3] = {bq, bk, bv};
  const float qscale = 1.4426950408889634f / 11.313708498984761f; // log2e/sqrt(128)

#pragma unroll
  for (int w = 0; w < 3; ++w) {
#pragma unroll
    for (int t = 0; t < 8; ++t) {
      v8f acc = {};
#pragma unroll
      for (int c = 0; c < 4; ++c)
        acc = wmma_bf16(xa[c], frag_B_bf16(Ws[w], CCH, t * 16, 32 * c, lr, hi), acc);
      const int col = t * 16 + lr;
      const float bb = bias[w][col];
#pragma unroll
      for (int i = 0; i < 8; ++i) {
        const int row = r0 + i + hi * 8;
        const float val = acc[i] + bb;
        if (w == 0)      qs  [(size_t)row * CCH + col] = (bf16)(val * qscale);
        else if (w == 1) kbuf[(size_t)row * CCH + col] = (bf16)val;
        else vT[((size_t)b * CCH + col) * NPOS + rb + i + hi * 8] = (bf16)val;
      }
    }
  }
}

// ---------------- Kernel 2: flash attention ---------------------------------
// 4 waves / block share double-buffered K/V tiles in LDS filled by async DMA.
__device__ __forceinline__ void issue_tile(const bf16* kg, const bf16* vg,
                                           bf16* ks, bf16* vs, int tid) {
#pragma unroll
  for (int j = 0; j < 4; ++j) {
    const int c = tid + j * 128;              // 16B chunk id, 0..511
    async_copy16(kg + c * 8, ks + c * 8);     // K tile: flat 8KB copy
    const int ch = c >> 2, sub = c & 3;       // V tile: 128 rows x 64B
    async_copy16(vg + (size_t)ch * NPOS + sub * 8, vs + ch * 32 + sub * 8);
  }
}

__global__ __launch_bounds__(128) void flash_attn(
    const float* __restrict__ x, const bf16* __restrict__ qs,
    const bf16* __restrict__ kbuf, const bf16* __restrict__ vT,
    float* __restrict__ out) {
  const int tid = threadIdx.x;
  const int lane = tid & 31, wave = tid >> 5;
  const int lr = lane & 15, hi = lane >> 4;
  const int tile = blockIdx.x * 4 + wave;     // 0..1023
  const int b = tile >> 8;                    // uniform across the block
  const int r0 = (tile & 255) * 16;

  const bf16* qb  = qs   + (size_t)b * NPOS * CCH;
  const bf16* kbp = kbuf + (size_t)b * NPOS * CCH;
  const bf16* vbp = vT   + (size_t)b * CCH * NPOS;

  v16bf qf[4];
#pragma unroll
  for (int c = 0; c < 4; ++c) qf[c] = frag_A_bf16(qb, CCH, r0, 32 * c, lr, hi);

  v8f o[8];
#pragma unroll
  for (int t = 0; t < 8; ++t) o[t] = (v8f){};
  float mrow[8], lrow[8];
#pragma unroll
  for (int i = 0; i < 8; ++i) { mrow[i] = -3.0e38f; lrow[i] = 0.0f; }

  __shared__ __align__(64) bf16 Ks[2][32][128];   // [buf][key][ch]   8KB each
  __shared__ __align__(64) bf16 Vs[2][128][32];   // [buf][ch][key]   8KB each
  __shared__ __align__(64) bf16 pbuf[4][16][32];  // per-wave P bounce
  bf16(*pw)[32] = pbuf[wave];

  issue_tile(kbp, vbp, &Ks[0][0][0], &Vs[0][0][0], tid);

  for (int it = 0; it < NT; ++it) {
    const int cur = it & 1, nxt = cur ^ 1;
    const bool more = (it + 1 < NT);
    if (more)
      issue_tile(kbp + (size_t)(it + 1) * 32 * CCH, vbp + (it + 1) * 32,
                 &Ks[nxt][0][0], &Vs[nxt][0][0], tid);
    if (more) tile_wait<8>(); else tile_wait<0>();
    __syncthreads();                              // buf[cur] resident everywhere

    const bf16* ksp = &Ks[cur][0][0];
    const bf16* vsp = &Vs[cur][0][0];

    // S = Q @ K^T (scale + log2e folded into Q)
    v8f s0 = {}, s1 = {};
#pragma unroll
    for (int c = 0; c < 4; ++c) {
      s0 = wmma_bf16(qf[c], frag_B_bf16(ksp, CCH, 0,  32 * c, lr, hi), s0);
      s1 = wmma_bf16(qf[c], frag_B_bf16(ksp, CCH, 16, 32 * c, lr, hi), s1);
    }
    // online softmax, row stats across 16-lane halves
    float alpha[8];
#pragma unroll
    for (int i = 0; i < 8; ++i) {
      float t = fmaxf(s0[i], s1[i]);
      t = fmaxf(t, __shfl_xor(t, 1, 32));
      t = fmaxf(t, __shfl_xor(t, 2, 32));
      t = fmaxf(t, __shfl_xor(t, 4, 32));
      t = fmaxf(t, __shfl_xor(t, 8, 32));
      const float mn = fmaxf(mrow[i], t);
      alpha[i] = exp2f(mrow[i] - mn);
      const float p0 = exp2f(s0[i] - mn);
      const float p1 = exp2f(s1[i] - mn);
      s0[i] = p0; s1[i] = p1;
      float rs = p0 + p1;
      rs += __shfl_xor(rs, 1, 32);
      rs += __shfl_xor(rs, 2, 32);
      rs += __shfl_xor(rs, 4, 32);
      rs += __shfl_xor(rs, 8, 32);
      lrow[i] = lrow[i] * alpha[i] + rs;
      mrow[i] = mn;
    }
#pragma unroll
    for (int t = 0; t < 8; ++t)
#pragma unroll
      for (int i = 0; i < 8; ++i) o[t][i] *= alpha[i];

    // C-layout -> A-layout bounce of P through wave-private LDS
#pragma unroll
    for (int i = 0; i < 8; ++i) {
      pw[i + hi * 8][lr]      = (bf16)s0[i];
      pw[i + hi * 8][lr + 16] = (bf16)s1[i];
    }
    asm volatile("s_wait_dscnt 0" ::: "memory");
    v16bf pf;
    {
      const bf16* r = &pw[lr][hi * 8];
      bf16x8 c0 = *(const bf16x8*)(r);
      bf16x8 c1 = *(const bf16x8*)(r + 16);
#pragma unroll
      for (int e = 0; e < 8; ++e) { pf[e] = c0[e]; pf[e + 8] = c1[e]; }
    }
    // O += P @ V
#pragma unroll
    for (int t = 0; t < 8; ++t)
      o[t] = wmma_bf16(pf, frag_B_bf16(vsp, 32, t * 16, 0, lr, hi), o[t]);

    __syncthreads();                              // done reading buf[cur]
  }

  // epilogue: out = x + O / l
  const float* xb = x + (size_t)b * NPOS * CCH;
  float* ob = out + (size_t)b * NPOS * CCH;
#pragma unroll
  for (int i = 0; i < 8; ++i) {
    const int row = r0 + i + hi * 8;
    const float inv = 1.0f / lrow[i];
#pragma unroll
    for (int t = 0; t < 8; ++t) {
      const int col = t * 16 + lr;
      const size_t idx = (size_t)row * CCH + col;
      ob[idx] = xb[idx] + o[t][i] * inv;
    }
  }
}

extern "C" void kernel_launch(void* const* d_in, const int* in_sizes, int n_in,
                              void* d_out, int out_size, void* d_ws, size_t ws_size,
                              hipStream_t stream) {
  (void)in_sizes; (void)n_in; (void)out_size; (void)ws_size;
  const float* x  = (const float*)d_in[0];
  const float* Wq = (const float*)d_in[1];
  const float* bq = (const float*)d_in[2];
  const float* Wk = (const float*)d_in[3];
  const float* bk = (const float*)d_in[4];
  const float* Wv = (const float*)d_in[5];
  const float* bv = (const float*)d_in[6];
  float* out = (float*)d_out;

  bf16* WqT = (bf16*)d_ws;
  bf16* WkT = WqT + CCH * CCH;
  bf16* WvT = WkT + CCH * CCH;
  bf16* qs  = WvT + CCH * CCH;
  bf16* kbf = qs  + (size_t)TOK * CCH;
  bf16* vT  = kbf + (size_t)TOK * CCH;

  prep_weights<<<192, 256, 0, stream>>>(Wq, Wk, Wv, WqT, WkT, WvT);
  qkv_proj   <<<256, 128, 0, stream>>>(x, bq, bk, bv, WqT, WkT, WvT, qs, kbf, vT);
  flash_attn <<<256, 128, 0, stream>>>(x, qs, kbf, vT, out);
}
